// MTPModule_51539607552186
// MI455X (gfx1250) — compile-verified, matrix operand order checked by reading, not run on hardware
//
#include <hip/hip_runtime.h>

// ---------------- problem constants ----------------
#define BATCH   4
#define SEQ     2048
#define NROW    (BATCH*SEQ)        // 8192 token rows
#define DMODEL  1024
#define DINNER  2048
#define DSTATE  128
#define NHEADS  32
#define HEADDIM 64
#define CONVDIM 2304               // DINNER + 2*DSTATE
#define DINPROJ 4384               // 2*DINNER + 2*DSTATE + NHEADS
#define NCHUNK  32
#define CHUNKLEN 64                // SEQ / NCHUNK

// ---------------- WMMA types ----------------
typedef __attribute__((ext_vector_type(16))) __bf16 v16bf;
typedef __attribute__((ext_vector_type(8)))  float  v8f;

__device__ __forceinline__ unsigned short f2bf(float f) {
  unsigned int u = __builtin_bit_cast(unsigned int, f);
  u += 0x7fffu + ((u >> 16) & 1u);            // round-to-nearest-even
  return (unsigned short)(u >> 16);
}
__device__ __forceinline__ float siluf(float x) {
  return x / (1.f + __expf(-x));
}

// ---------------- f32 -> bf16 weight conversion ----------------
__global__ __launch_bounds__(256)
void cvt_bf16_kernel(const float* __restrict__ in, unsigned short* __restrict__ out, int n) {
  int i = blockIdx.x * 256 + threadIdx.x;
  if (i < n) out[i] = f2bf(in[i]);
}

// ---------------- RMSNorm(h_prev) ++ RMSNorm(emb_next) -> bf16 (NROW x 2048) ----------------
__global__ __launch_bounds__(256)
void rmsnorm_concat_kernel(const float* __restrict__ h_prev, const float* __restrict__ emb,
                           const float* __restrict__ w, unsigned short* __restrict__ out) {
  int r = blockIdx.x;
  int t = threadIdx.x;
  __shared__ float red[8];
  for (int half = 0; half < 2; ++half) {
    const float* src = (half ? emb : h_prev) + (size_t)r * DMODEL;
    float v[4]; float ss = 0.f;
    #pragma unroll
    for (int i = 0; i < 4; ++i) { v[i] = src[t + i * 256]; ss += v[i] * v[i]; }
    #pragma unroll
    for (int off = 16; off > 0; off >>= 1) ss += __shfl_down(ss, off);
    if ((t & 31) == 0) red[t >> 5] = ss;
    __syncthreads();
    if (t == 0) {
      float tot = 0.f;
      #pragma unroll
      for (int i = 0; i < 8; ++i) tot += red[i];
      red[0] = rsqrtf(tot * (1.f / DMODEL) + 1e-6f);
    }
    __syncthreads();
    float scale = red[0];
    #pragma unroll
    for (int i = 0; i < 4; ++i) {
      int c = t + i * 256;
      out[(size_t)r * (2 * DMODEL) + half * DMODEL + c] = f2bf(w[c] * v[i] * scale);
    }
    __syncthreads();
  }
}

// ---------------- bf16 WMMA GEMM: C(M,N) = A(M,K) * W(N,K)^T (+bias) ----------------
// 128x128 tile / 256 threads (8 waves, each wave a 32x64 sub-tile = 8 WMMA frags).
// Global->LDS staging is register-pipelined: tile k+1's 4x global_load_b128 are
// issued right after the barrier and land while tile k's 8 WMMAs execute.
#define BM 128
#define BN 128
#define BK 32
#define LSTR 40   // padded LDS row stride in bf16 units (80B -> conflict-free b128 frag reads)

union V16U { v16bf v; uint4 u[2]; };

__global__ __launch_bounds__(256)
void gemm_bf16_kernel(const unsigned short* __restrict__ A,
                      const unsigned short* __restrict__ W,
                      const float* __restrict__ bias,
                      float* __restrict__ outF,
                      unsigned short* __restrict__ outB,
                      int M, int N, int K) {
  __shared__ __align__(16) unsigned short As[BM * LSTR];
  __shared__ __align__(16) unsigned short Ws[BN * LSTR];
  const int tid  = threadIdx.x;
  const int bm   = blockIdx.y * BM;
  const int bn   = blockIdx.x * BN;
  const int wave = tid >> 5, lane = tid & 31;
  const int wr   = (wave >> 1) * 32;   // wave row offset in tile
  const int wc   = (wave & 1) * 64;    // wave col offset in tile
  const int rA   = lane & 15;
  const int koff = (lane & 16) ? 8 : 0; // per-lane K swizzle for 16-bit A/B fragments

  // ---- staging assignment: thread owns 4 fixed b128 chunks per K-tile ----
  const int r0 = tid >> 2;             // 0..63
  const int hf = (tid & 3) * 8;        // 0,8,16,24 (bf16 units)
  const unsigned short* gA0 = A + (size_t)(bm + r0) * K + hf;
  const unsigned short* gA1 = A + (size_t)(bm + r0 + 64) * K + hf;
  int wRow0 = bn + r0;      if (wRow0 >= N) wRow0 = 0;   // clamp; masked at store
  int wRow1 = bn + r0 + 64; if (wRow1 >= N) wRow1 = 0;
  const unsigned short* gW0 = W + (size_t)wRow0 * K + hf;
  const unsigned short* gW1 = W + (size_t)wRow1 * K + hf;
  unsigned short* sA0 = &As[r0 * LSTR + hf];
  unsigned short* sA1 = &As[(r0 + 64) * LSTR + hf];
  unsigned short* sW0 = &Ws[r0 * LSTR + hf];
  unsigned short* sW1 = &Ws[(r0 + 64) * LSTR + hf];

  v8f acc[2][4];
  #pragma unroll
  for (int i = 0; i < 2; ++i)
    #pragma unroll
    for (int j = 0; j < 4; ++j) acc[i][j] = {};

  // prefetch first tile into registers
  uint4 ra0 = *(const uint4*)gA0;
  uint4 ra1 = *(const uint4*)gA1;
  uint4 rw0 = *(const uint4*)gW0;
  uint4 rw1 = *(const uint4*)gW1;

  const int nk = K / BK;
  for (int kt = 0; kt < nk; ++kt) {
    __syncthreads();                   // previous tile's frag reads complete
    *(uint4*)sA0 = ra0; *(uint4*)sA1 = ra1;
    *(uint4*)sW0 = rw0; *(uint4*)sW1 = rw1;
    __syncthreads();
    if (kt + 1 < nk) {                 // issue next tile's loads; land during WMMAs
      gA0 += BK; gA1 += BK; gW0 += BK; gW1 += BK;
      ra0 = *(const uint4*)gA0; ra1 = *(const uint4*)gA1;
      rw0 = *(const uint4*)gW0; rw1 = *(const uint4*)gW1;
    }

    v16bf af[2], wf[4];
    #pragma unroll
    for (int i = 0; i < 2; ++i) {
      int row = wr + i * 16 + rA;
      V16U u; u.u[0] = *(const uint4*)&As[row * LSTR + koff];
              u.u[1] = *(const uint4*)&As[row * LSTR + koff + 16];
      af[i] = u.v;
    }
    #pragma unroll
    for (int j = 0; j < 4; ++j) {
      int row = wc + j * 16 + rA;
      V16U u; u.u[0] = *(const uint4*)&Ws[row * LSTR + koff];
              u.u[1] = *(const uint4*)&Ws[row * LSTR + koff + 16];
      wf[j] = u.v;
    }
    #pragma unroll
    for (int i = 0; i < 2; ++i)
      #pragma unroll
      for (int j = 0; j < 4; ++j)
        acc[i][j] = __builtin_amdgcn_wmma_f32_16x16x32_bf16(
            false, af[i], false, wf[j], (short)0, acc[i][j], false, false);
  }

  // epilogue: C frag VGPR e holds M=e (lanes 0-15) / M=8+e (lanes 16-31), N=lane&15
  const int rsel = (lane >> 4) * 8;
  #pragma unroll
  for (int i = 0; i < 2; ++i) {
    #pragma unroll
    for (int j = 0; j < 4; ++j) {
      int col = bn + wc + j * 16 + (lane & 15);
      float bv = (bias && col < N) ? bias[col] : 0.f;
      #pragma unroll
      for (int e = 0; e < 8; ++e) {
        int row = bm + wr + i * 16 + rsel + e;
        if (col < N) {
          float v = acc[i][j][e] + bv;
          size_t idx = (size_t)row * N + col;
          if (outF) outF[idx] = v;
          if (outB) outB[idx] = f2bf(v);
        }
      }
    }
  }
}

// ---------------- depthwise causal conv(4) + SiLU on xBC channels ----------------
__global__ __launch_bounds__(256)
void conv_silu_kernel(const float* __restrict__ zx, const float* __restrict__ cw,
                      const float* __restrict__ cb, float* __restrict__ out) {
  int idx = blockIdx.x * 256 + threadIdx.x;
  if (idx >= NROW * CONVDIM) return;
  int c  = idx % CONVDIM;
  int bl = idx / CONVDIM;
  int l  = bl % SEQ, b = bl / SEQ;
  float acc = cb[c];
  const float* w4 = cw + c * 4;
  #pragma unroll
  for (int j = 0; j < 4; ++j) {
    int ll = l - 3 + j;
    if (ll >= 0)
      acc += w4[j] * zx[((size_t)b * SEQ + ll) * DINPROJ + DINNER + c];
  }
  out[(size_t)bl * CONVDIM + c] = siluf(acc);
}

// ---------------- dt = softplus(dt_raw + bias); dA = exp(dt * -exp(A_log)) ----------------
__global__ __launch_bounds__(256)
void dt_kernel(const float* __restrict__ zx, const float* __restrict__ dt_bias,
               const float* __restrict__ A_log,
               float* __restrict__ dt_out, float* __restrict__ dA_out) {
  int idx = blockIdx.x * 256 + threadIdx.x;
  if (idx >= NROW * NHEADS) return;
  int h = idx & (NHEADS - 1);
  int bl = idx >> 5;
  float x = zx[(size_t)bl * DINPROJ + (DINPROJ - NHEADS) + h] + dt_bias[h];
  float dt = (x > 20.f) ? x : log1pf(__expf(x));
  dt_out[idx] = dt;
  dA_out[idx] = __expf(dt * (-__expf(A_log[h])));
}

// ---------------- chunked selective scan ----------------
// block = 256 threads: thread t owns state[p= t>>2][n = (t&3)*32 .. +32)
template <bool WRITE_Y>
__global__ __launch_bounds__(256)
void scan_chunk_kernel(const float* __restrict__ xbc, const float* __restrict__ dt_arr,
                       const float* __restrict__ dA_arr, float* __restrict__ cstate,
                       float* __restrict__ cP, const float* __restrict__ Dp,
                       float* __restrict__ y) {
  int cid = blockIdx.x;                    // ((b*32 + h)*32 + chunk)
  int chunk = cid & (NCHUNK - 1);
  int h = (cid >> 5) & (NHEADS - 1);
  int b = cid >> 10;
  int t = threadIdx.x, p = t >> 2, nb = (t & 3) * 32;

  float s[32];
  if (WRITE_Y) {
    size_t base = (size_t)cid * (HEADDIM * DSTATE) + (size_t)t * 32;
    #pragma unroll
    for (int j = 0; j < 32; ++j) s[j] = cstate[base + j];   // corrected init state
  } else {
    #pragma unroll
    for (int j = 0; j < 32; ++j) s[j] = 0.f;
  }
  float P = 1.f;
  float Dh = Dp[h];
  __shared__ float sx[HEADDIM], sB[DSTATE], sC[DSTATE];

  for (int li = 0; li < CHUNKLEN; ++li) {
    int l = chunk * CHUNKLEN + li;
    size_t bl = (size_t)b * SEQ + l;
    const float* row = xbc + bl * CONVDIM;
    __syncthreads();
    if (t < HEADDIM) sx[t] = row[h * HEADDIM + t];
    if (t < DSTATE)  sB[t] = row[DINNER + t];
    else             sC[t - DSTATE] = row[DINNER + DSTATE + (t - DSTATE)];
    __syncthreads();
    float dtv = dt_arr[bl * NHEADS + h];
    float dAv = dA_arr[bl * NHEADS + h];
    if (!WRITE_Y) P *= dAv;
    float coef = dtv * sx[p];
    float acc = 0.f;
    #pragma unroll
    for (int j = 0; j < 32; ++j) {
      s[j] = s[j] * dAv + coef * sB[nb + j];
      if (WRITE_Y) acc += s[j] * sC[nb + j];
    }
    if (WRITE_Y) {
      acc += __shfl_xor(acc, 1);
      acc += __shfl_xor(acc, 2);
      if ((t & 3) == 0)
        y[bl * DINNER + h * HEADDIM + p] = acc + Dh * sx[p];
    }
  }
  if (!WRITE_Y) {
    size_t base = (size_t)cid * (HEADDIM * DSTATE) + (size_t)t * 32;
    #pragma unroll
    for (int j = 0; j < 32; ++j) cstate[base + j] = s[j];
    if (t == 0) cP[cid] = P;
  }
}

// prefix over chunk states: S_init[c] written in place; run = run*P[c] + S_local[c]
__global__ __launch_bounds__(256)
void scan_prefix_kernel(float* __restrict__ cstate, const float* __restrict__ cP) {
  int bh = blockIdx.x;             // 0..127
  int t = threadIdx.x;
  float run[32];
  #pragma unroll
  for (int j = 0; j < 32; ++j) run[j] = 0.f;
  for (int c = 0; c < NCHUNK; ++c) {
    size_t base = ((size_t)bh * NCHUNK + c) * (HEADDIM * DSTATE) + (size_t)t * 32;
    float P = cP[bh * NCHUNK + c];
    #pragma unroll
    for (int j = 0; j < 32; ++j) {
      float loc = cstate[base + j];
      cstate[base + j] = run[j];         // becomes corrected init state for chunk c
      run[j] = run[j] * P + loc;
    }
  }
}

// ---------------- gated RMSNorm: y*silu(z), rmsnorm, -> bf16 ----------------
__global__ __launch_bounds__(256)
void gated_norm_kernel(const float* __restrict__ y, const float* __restrict__ zx,
                       const float* __restrict__ gw, unsigned short* __restrict__ out) {
  int r = blockIdx.x;
  int t = threadIdx.x;
  __shared__ float red[8];
  float v[8]; float ss = 0.f;
  #pragma unroll
  for (int i = 0; i < 8; ++i) {
    int c = t + i * 256;
    float z = zx[(size_t)r * DINPROJ + c];
    float yy = y[(size_t)r * DINNER + c] * siluf(z);
    v[i] = yy; ss += yy * yy;
  }
  #pragma unroll
  for (int off = 16; off > 0; off >>= 1) ss += __shfl_down(ss, off);
  if ((t & 31) == 0) red[t >> 5] = ss;
  __syncthreads();
  if (t == 0) {
    float tot = 0.f;
    #pragma unroll
    for (int i = 0; i < 8; ++i) tot += red[i];
    red[0] = rsqrtf(tot * (1.f / DINNER) + 1e-5f);
  }
  __syncthreads();
  float scale = red[0];
  #pragma unroll
  for (int i = 0; i < 8; ++i) {
    int c = t + i * 256;
    out[(size_t)r * DINNER + c] = f2bf(gw[c] * v[i] * scale);
  }
}

// ---------------- decode: out(r,3) = h_k(r,:) . dec_W(3,1024)^T + dec_b ----------------
__global__ __launch_bounds__(256)
void dec_kernel(const float* __restrict__ hk, const float* __restrict__ decW,
                const float* __restrict__ decb, float* __restrict__ out) {
  int r = blockIdx.x;
  int t = threadIdx.x;
  float a0 = 0.f, a1 = 0.f, a2 = 0.f;
  for (int i = t; i < DMODEL; i += 256) {
    float h = hk[(size_t)r * DMODEL + i];
    a0 += h * decW[i];
    a1 += h * decW[DMODEL + i];
    a2 += h * decW[2 * DMODEL + i];
  }
  #pragma unroll
  for (int off = 16; off > 0; off >>= 1) {
    a0 += __shfl_down(a0, off); a1 += __shfl_down(a1, off); a2 += __shfl_down(a2, off);
  }
  __shared__ float red[8][3];
  if ((t & 31) == 0) { int w = t >> 5; red[w][0] = a0; red[w][1] = a1; red[w][2] = a2; }
  __syncthreads();
  if (t == 0) {
    float s0 = decb[0], s1 = decb[1], s2 = decb[2];
    #pragma unroll
    for (int w = 0; w < 8; ++w) { s0 += red[w][0]; s1 += red[w][1]; s2 += red[w][2]; }
    out[(size_t)r * 3 + 0] = s0;
    out[(size_t)r * 3 + 1] = s1;
    out[(size_t)r * 3 + 2] = s2;
  }
}

// ---------------- host launcher ----------------
extern "C" void kernel_launch(void* const* d_in, const int* in_sizes, int n_in,
                              void* d_out, int out_size, void* d_ws, size_t ws_size,
                              hipStream_t stream) {
  (void)in_sizes; (void)n_in; (void)out_size; (void)ws_size;
  const float* h_prev   = (const float*)d_in[0];
  const float* emb_next = (const float*)d_in[1];
  const float* norm_w   = (const float*)d_in[2];
  const float* proj_W   = (const float*)d_in[3];
  const float* proj_b   = (const float*)d_in[4];
  const float* in_proj_W= (const float*)d_in[5];
  const float* conv_w   = (const float*)d_in[6];
  const float* conv_b   = (const float*)d_in[7];
  const float* dt_bias  = (const float*)d_in[8];
  const float* A_log    = (const float*)d_in[9];
  const float* D_param  = (const float*)d_in[10];
  const float* gnorm_w  = (const float*)d_in[11];
  const float* out_proj_W = (const float*)d_in[12];
  const float* dec_W    = (const float*)d_in[13];
  const float* dec_b    = (const float*)d_in[14];

  float* out_head = (float*)d_out;                       // (8192,3)
  float* out_hk   = out_head + (size_t)NROW * 3;         // (8192,1024)

  char* ws = (char*)d_ws;
  size_t off = 0;
  auto take = [&](size_t bytes) -> void* {
    void* p = ws + off; off = (off + bytes + 255) & ~(size_t)255; return p;
  };
  unsigned short* combined_b = (unsigned short*)take((size_t)NROW * 2 * DMODEL * 2);
  unsigned short* projW_b    = (unsigned short*)take((size_t)DMODEL * 2 * DMODEL * 2);
  unsigned short* inprojW_b  = (unsigned short*)take((size_t)DINPROJ * DMODEL * 2);
  unsigned short* outprojW_b = (unsigned short*)take((size_t)DMODEL * DINNER * 2);
  unsigned short* hprime_b   = (unsigned short*)take((size_t)NROW * DMODEL * 2);
  float* zx    = (float*)take((size_t)NROW * DINPROJ * 4);
  float* xbc   = (float*)take((size_t)NROW * CONVDIM * 4);
  float* dt_a  = (float*)take((size_t)NROW * NHEADS * 4);
  float* dA_a  = (float*)take((size_t)NROW * NHEADS * 4);
  float* y_buf = (float*)take((size_t)NROW * DINNER * 4);
  unsigned short* ynorm_b = (unsigned short*)take((size_t)NROW * DINNER * 2);
  float* cstate = (float*)take((size_t)BATCH * NHEADS * NCHUNK * HEADDIM * DSTATE * 4);
  float* cP     = (float*)take((size_t)BATCH * NHEADS * NCHUNK * 4);

  // weights -> bf16
  {
    int n1 = DMODEL * 2 * DMODEL, n2 = DINPROJ * DMODEL, n3 = DMODEL * DINNER;
    cvt_bf16_kernel<<<(n1 + 255) / 256, 256, 0, stream>>>(proj_W, projW_b, n1);
    cvt_bf16_kernel<<<(n2 + 255) / 256, 256, 0, stream>>>(in_proj_W, inprojW_b, n2);
    cvt_bf16_kernel<<<(n3 + 255) / 256, 256, 0, stream>>>(out_proj_W, outprojW_b, n3);
  }
  // RMSNorm + concat
  rmsnorm_concat_kernel<<<NROW, 256, 0, stream>>>(h_prev, emb_next, norm_w, combined_b);
  // GEMM1: h_prime = combined @ proj_W^T + proj_b  (bf16 out)
  gemm_bf16_kernel<<<dim3((DMODEL + BN - 1) / BN, NROW / BM), 256, 0, stream>>>(
      combined_b, projW_b, proj_b, nullptr, hprime_b, NROW, DMODEL, 2 * DMODEL);
  // GEMM2: zxbcdt = h_prime @ in_proj_W^T  (f32 out)
  gemm_bf16_kernel<<<dim3((DINPROJ + BN - 1) / BN, NROW / BM), 256, 0, stream>>>(
      hprime_b, inprojW_b, nullptr, zx, nullptr, NROW, DINPROJ, DMODEL);
  // conv + silu, dt/dA
  conv_silu_kernel<<<(NROW * CONVDIM + 255) / 256, 256, 0, stream>>>(zx, conv_w, conv_b, xbc);
  dt_kernel<<<(NROW * NHEADS + 255) / 256, 256, 0, stream>>>(zx, dt_bias, A_log, dt_a, dA_a);
  // chunked scan: local -> prefix -> output
  scan_chunk_kernel<false><<<BATCH * NHEADS * NCHUNK, 256, 0, stream>>>(
      xbc, dt_a, dA_a, cstate, cP, D_param, y_buf);
  scan_prefix_kernel<<<BATCH * NHEADS, 256, 0, stream>>>(cstate, cP);
  scan_chunk_kernel<true><<<BATCH * NHEADS * NCHUNK, 256, 0, stream>>>(
      xbc, dt_a, dA_a, cstate, cP, D_param, y_buf);
  // gated RMSNorm -> bf16
  gated_norm_kernel<<<NROW, 256, 0, stream>>>(y_buf, zx, gnorm_w, ynorm_b);
  // GEMM3: h_k = y @ out_proj_W^T  (f32 straight into d_out)
  gemm_bf16_kernel<<<dim3((DMODEL + BN - 1) / BN, NROW / BM), 256, 0, stream>>>(
      ynorm_b, outprojW_b, nullptr, out_hk, nullptr, NROW, DMODEL, DINNER);
  // decode head
  dec_kernel<<<NROW, 256, 0, stream>>>(out_hk, dec_W, dec_b, out_head);
}
